// NN_ECS_8340826489063
// MI455X (gfx1250) — matrix operand order repaired and checked by reading, not compiled, not run on hardware
//
#include <hip/hip_runtime.h>
#include <math.h>

typedef __attribute__((ext_vector_type(16))) _Float16 v16h;
typedef __attribute__((ext_vector_type(8)))  float    v8f;

#define NGRAPH 1024
#define NPG    100
#define DEG    32
#define EPG    (NPG*DEG)      // 3200 edges per graph
#define HF     16
#define TILES  (EPG/16)       // 200 tiles of 16 edges
#define NWAVE  8
#define BLK    256

struct FPParams {
  const float *pn_W, *pn_b;     // [12,16],[16]
  const float *e1_W, *e1_b;     // [17,16],[16]
  const float *e2_W, *e2_b;     // [32,1],[1]
  const float *et_W, *et_b;     // [16,16],[16]
  const float *g1_Wih, *g1_bih, *g1_Whh, *g1_bhh;  // [48,16],[48]
  const float *cl_W, *cl_b;     // [32,1],[1]
  const float *pno_W, *pno_b;   // [16,16],[16]
  const float *g2_Wih, *g2_bih, *g2_Whh, *g2_bhh;
};

struct HeadParams {
  const float *fc1_W,*fc1_b,*fc2_W,*fc2_b,*fc3_W,*fc3_b,
              *fc4_W,*fc4_b,*fc5_W,*fc5_b,*fcf_W,*fcf_b;
};

__device__ __forceinline__ float leakyf(float x){ return x >= 0.f ? x : 0.01f*x; }
__device__ __forceinline__ float sigmf (float x){ return 1.f/(1.f+expf(-x)); }
__device__ __forceinline__ float eluf  (float x){ return x > 0.f ? x : (expf(x)-1.f); }

// 16-dim GRU cell: gi = xin@Wih^T+bih ; gh = h@Whh^T+bhh ; standard r/z/n gates.
__device__ void gru16(const float* xin, const float* h,
                      const float* __restrict__ Wih, const float* __restrict__ bih,
                      const float* __restrict__ Whh, const float* __restrict__ bhh,
                      float* out)
{
  float gi[48], gh[48];
  #pragma unroll
  for (int j=0;j<48;j++){ gi[j]=bih[j]; gh[j]=bhh[j]; }
  for (int k=0;k<16;k++){
    float xv=xin[k], hv=h[k];
    #pragma unroll
    for (int j=0;j<48;j++){ gi[j]+=xv*Wih[j*16+k]; gh[j]+=hv*Whh[j*16+k]; }
  }
  #pragma unroll
  for (int f=0;f<16;f++){
    float r=sigmf(gi[f]+gh[f]);
    float z=sigmf(gi[16+f]+gh[16+f]);
    float n=tanhf(gi[32+f]+r*gh[32+f]);
    out[f]=(1.f-z)*n + z*h[f];
  }
}

// One workgroup per graph; whole AttentiveFP layer + readout fused in LDS.
__global__ __launch_bounds__(BLK) void afp_graph_kernel(
    const int* __restrict__ src, const int* __restrict__ dst,
    const float* __restrict__ nf, const float* __restrict__ ef,
    FPParams P, float* __restrict__ outg)
{
  __shared__ float s_nf [NPG][12];
  __shared__ float s_hv [NPG][HF];
  __shared__ float s_ctx[NPG][HF];
  __shared__ float s_h  [NPG][HF];
  __shared__ float s_exps[NPG];            // edge-softmax denom per dst node; reused for node softmax
  __shared__ float s_tile[NWAVE][16][HF];  // he1 tile per wave
  __shared__ float s_a  [NWAVE][16];
  __shared__ int   s_dl [NWAVE][16];
  __shared__ float s_w2[32];
  __shared__ float s_g [HF];
  __shared__ float s_gr[HF];
  __shared__ float s_zsum;

  const int g    = blockIdx.x;
  const int tid  = threadIdx.x;
  const int lane = tid & 31;
  const int wv   = tid >> 5;
  const long edgeBase = (long)g * EPG;
  const long nodeBase = (long)g * NPG;

  // ---- phase 0: stage node features, clear accumulators ----
  for (int i = tid; i < NPG*12; i += BLK) s_nf[i/12][i%12] = nf[nodeBase*12 + i];
  for (int i = tid; i < NPG*HF; i += BLK) s_ctx[i/HF][i%HF] = 0.f;
  for (int i = tid; i < NPG;    i += BLK) s_exps[i] = 0.f;
  if (tid < 32) s_w2[tid] = P.e2_W[tid];
  if (tid == 0) s_zsum = 0.f;
  for (int i = tid; i < HF; i += BLK) s_gr[i] = 0.f;
  __syncthreads();

  // ---- node projection: hv = leaky(nf @ pn_W + b) ----
  if (tid < NPG) {
    float o[HF];
    #pragma unroll
    for (int f=0;f<HF;f++) o[f]=P.pn_b[f];
    for (int k=0;k<12;k++){ float v=s_nf[tid][k];
      #pragma unroll
      for (int f=0;f<HF;f++) o[f]+=v*P.pn_W[k*HF+f]; }
    #pragma unroll
    for (int f=0;f<HF;f++) s_hv[tid][f]=leakyf(o[f]);
  }

  // ---- pack WMMA B operands (same K order as A: lanes0-15 hold K0..7,K16..23 of col n,
  //      lanes16-31 hold K8..15,K24..31). e1_W is 17x16 (K pad 32), et_W 16x16 (K pad 32).
  const int col = lane & 15;
  v16h b1, b2;
  #pragma unroll
  for (int j=0;j<16;j++){ b1[j]=(_Float16)0.f; b2[j]=(_Float16)0.f; }
  if (lane < 16) {
    #pragma unroll
    for (int k=0;k<8;k++){ b1[k]=(_Float16)P.e1_W[k*HF+col]; b2[k]=(_Float16)P.et_W[k*HF+col]; }
    b1[8]=(_Float16)P.e1_W[16*HF+col];   // K=16 row of e1_W
  } else {
    #pragma unroll
    for (int k=0;k<8;k++){ b1[k]=(_Float16)P.e1_W[(8+k)*HF+col]; b2[k]=(_Float16)P.et_W[(8+k)*HF+col]; }
  }
  const float cb1 = P.e1_b[col];
  const float cb2 = P.et_b[col];
  const float e2b = P.e2_b[0];
  const int   m   = lane & 15;           // matrix row (edge within tile) this lane feeds
  const int rbase = (lane < 16) ? 0 : 8; // C/D layout: vgpr r -> row r (lanes<16) / r+8
  __syncthreads();

  // he1 tile via WMMA: A layout (16-bit, 16x32): lanes0-15 halves = K0..7,K16..23;
  // lanes16-31 halves = K8..15,K24..31. concat([nf(12), ef(5)]) -> K0..16, rest 0.
  auto computeHE1 = [&](int t) -> v8f {
    int e  = t*16 + m;
    int sl = src[edgeBase + e] - g*NPG;
    const float* efp = &ef[(edgeBase + e)*5];
    v16h a;
    #pragma unroll
    for (int j=0;j<16;j++) a[j]=(_Float16)0.f;
    if (lane < 16) {
      #pragma unroll
      for (int k=0;k<8;k++) a[k]=(_Float16)s_nf[sl][k];     // K0..7
      a[8]=(_Float16)efp[4];                                 // K16 = ef[4]
    } else {
      #pragma unroll
      for (int k=0;k<4;k++) a[k]  =(_Float16)s_nf[sl][8+k];  // K8..11
      #pragma unroll
      for (int k=0;k<4;k++) a[4+k]=(_Float16)efp[k];         // K12..15 = ef[0..3]
    }
    v8f c;
    #pragma unroll
    for (int r=0;r<8;r++) c[r]=cb1;
    c = __builtin_amdgcn_wmma_f32_16x16x32_f16(false, a, false, b1, (short)0, c, false, false);
    #pragma unroll
    for (int r=0;r<8;r++) c[r]=leakyf(c[r]);
    return c;
  };

  // ---- pass A: logits + per-dst exp sums (softmax denom; no max-sub needed, logits O(1)) ----
  for (int t = wv; t < TILES; t += NWAVE) {
    v8f c1 = computeHE1(t);
    #pragma unroll
    for (int r=0;r<8;r++) s_tile[wv][rbase+r][col] = c1[r];
    __syncthreads();
    if (lane < 16) {
      int e  = t*16 + lane;
      int dl = dst[edgeBase + e] - g*NPG;
      float acc = e2b;
      #pragma unroll
      for (int k=0;k<16;k++) acc += s_hv[dl][k]*s_w2[k];
      #pragma unroll
      for (int k=0;k<16;k++) acc += s_tile[wv][lane][k]*s_w2[16+k];
      atomicAdd(&s_exps[dl], expf(leakyf(acc)));
    }
    __syncthreads();
  }

  // ---- pass B: recompute he1, normalize, msg = a*(he1@et_W+b), LDS-atomic scatter ----
  for (int t = wv; t < TILES; t += NWAVE) {
    v8f c1 = computeHE1(t);
    #pragma unroll
    for (int r=0;r<8;r++) s_tile[wv][rbase+r][col] = c1[r];
    __syncthreads();
    if (lane < 16) {
      int e  = t*16 + lane;
      int dl = dst[edgeBase + e] - g*NPG;
      float acc = e2b;
      #pragma unroll
      for (int k=0;k<16;k++) acc += s_hv[dl][k]*s_w2[k];
      #pragma unroll
      for (int k=0;k<16;k++) acc += s_tile[wv][lane][k]*s_w2[16+k];
      s_a [wv][lane] = expf(leakyf(acc)) / s_exps[dl];
      s_dl[wv][lane] = dl;
    }
    __syncthreads();
    v16h a2;
    #pragma unroll
    for (int j=0;j<16;j++) a2[j]=(_Float16)0.f;
    if (lane < 16) {
      #pragma unroll
      for (int k=0;k<8;k++) a2[k]=(_Float16)s_tile[wv][m][k];    // K0..7
    } else {
      #pragma unroll
      for (int k=0;k<8;k++) a2[k]=(_Float16)s_tile[wv][m][8+k];  // K8..15
    }
    v8f c2;
    #pragma unroll
    for (int r=0;r<8;r++) c2[r]=cb2;
    c2 = __builtin_amdgcn_wmma_f32_16x16x32_f16(false, a2, false, b2, (short)0, c2, false, false);
    #pragma unroll
    for (int r=0;r<8;r++) {
      int row = rbase + r;
      atomicAdd(&s_ctx[s_dl[wv][row]][col], s_a[wv][row]*c2[r]);
    }
    __syncthreads();
  }

  // ---- ctx = elu(sum) ----
  for (int i = tid; i < NPG*HF; i += BLK) {
    float v = s_ctx[i/HF][i%HF];
    s_ctx[i/HF][i%HF] = eluf(v);
  }
  __syncthreads();

  // ---- GRU1 + relu per node ----
  if (tid < NPG) {
    float hn[HF];
    gru16(&s_ctx[tid][0], &s_hv[tid][0], P.g1_Wih, P.g1_bih, P.g1_Whh, P.g1_bhh, hn);
    #pragma unroll
    for (int f=0;f<HF;f++) s_h[tid][f] = fmaxf(hn[f], 0.f);
  }
  __syncthreads();

  // ---- readout: g = segment_sum(h) ----
  if (tid < HF) {
    float s = 0.f;
    for (int n=0;n<NPG;n++) s += s_h[n][tid];
    s_g[tid] = s;
  }
  if (tid == 0) { }  // s_zsum already 0
  __syncthreads();

  // node attention logits z, softmax over nodes
  if (tid < NPG) {
    float acc = P.cl_b[0];
    #pragma unroll
    for (int k=0;k<16;k++) acc += fmaxf(s_g[k],0.f)*P.cl_W[k];
    #pragma unroll
    for (int k=0;k<16;k++) acc += s_h[tid][k]*P.cl_W[16+k];
    float ex = expf(leakyf(acc));
    s_exps[tid] = ex;                 // reuse as node-exp store
    atomicAdd(&s_zsum, ex);
  }
  __syncthreads();

  // g_repr = sum an * (h @ pno_W + b)
  if (tid < NPG) {
    float an = s_exps[tid] / s_zsum;
    float pv[HF];
    #pragma unroll
    for (int f=0;f<HF;f++) pv[f]=P.pno_b[f];
    for (int k=0;k<16;k++){ float hv=s_h[tid][k];
      #pragma unroll
      for (int f=0;f<HF;f++) pv[f]+=hv*P.pno_W[k*HF+f]; }
    #pragma unroll
    for (int f=0;f<HF;f++) atomicAdd(&s_gr[f], an*pv[f]);
  }
  __syncthreads();

  // GRU2 on [1,16] + relu -> graph embedding
  if (tid == 0) {
    float xin[HF], go[HF];
    #pragma unroll
    for (int f=0;f<HF;f++) xin[f]=eluf(s_gr[f]);
    gru16(xin, s_g, P.g2_Wih, P.g2_bih, P.g2_Whh, P.g2_bhh, go);
    #pragma unroll
    for (int f=0;f<HF;f++) outg[(long)g*HF + f] = fmaxf(go[f], 0.f);
  }
}

// Cosine-combine of the two graph embeddings + residual MLP head. One thread per graph.
__global__ __launch_bounds__(256) void head_kernel(
    const float* __restrict__ h1, const float* __restrict__ h2,
    const float* __restrict__ x, HeadParams P, float* __restrict__ out)
{
  int b = blockIdx.x*blockDim.x + threadIdx.x;
  if (b >= NGRAPH) return;
  float a1[HF], a2[HF], n1=0.f, n2=0.f;
  #pragma unroll
  for (int f=0;f<HF;f++){ a1[f]=h1[b*HF+f]; n1+=a1[f]*a1[f];
                          a2[f]=h2[b*HF+f]; n2+=a2[f]*a2[f]; }
  float inv = 1.f/(sqrtf(n1)*sqrtf(n2));
  float hh[32];
  #pragma unroll
  for (int f=0;f<HF;f++){ hh[f]=a1[f]*a2[f]*inv; hh[16+f]=1.f; }

  float x0=x[b*2], x1=x[b*2+1];
  float o1[32], o3[32], o5[32], o7[32], o8[32];
  #pragma unroll
  for (int j=0;j<32;j++) o1[j]=P.fc1_b[j]+x0*P.fc1_W[j]+x1*P.fc1_W[32+j];
  for (int j=0;j<32;j++){ float acc=P.fc2_b[j];
    for (int i=0;i<32;i++) acc+=o1[i]*P.fc2_W[i*32+j];
    o3[j]=tanhf(acc); }
  for (int j=0;j<32;j++){ float acc=P.fc3_b[j]+o1[j];
    for (int i=0;i<32;i++) acc+=o3[i]*P.fc3_W[i*32+j];
    o5[j]=tanhf(acc); }
  for (int j=0;j<32;j++){ float acc=P.fc4_b[j];
    for (int i=0;i<32;i++) acc+=o5[i]*P.fc4_W[i*32+j];
    o7[j]=tanhf(acc); }
  for (int j=0;j<32;j++){ float acc=P.fc5_b[j]+o5[j];
    for (int i=0;i<32;i++) acc+=o7[i]*P.fc5_W[i*32+j];
    o8[j]=acc; }
  float f0=P.fcf_b[0], f1=P.fcf_b[1];
  #pragma unroll
  for (int i=0;i<32;i++){ float v=hh[i]*o8[i]; f0+=v*P.fcf_W[i*2]; f1+=v*P.fcf_W[i*2+1]; }
  out[b*2+0]=f0; out[b*2+1]=f1;
}

extern "C" void kernel_launch(void* const* d_in, const int* in_sizes, int n_in,
                              void* d_out, int out_size, void* d_ws, size_t ws_size,
                              hipStream_t stream) {
  (void)in_sizes; (void)n_in; (void)out_size; (void)ws_size;
  const int*   src1 = (const int*)  d_in[0];
  const int*   dst1 = (const int*)  d_in[1];
  const float* nf1  = (const float*)d_in[2];
  const float* ef1  = (const float*)d_in[3];
  const int*   src2 = (const int*)  d_in[4];
  const int*   dst2 = (const int*)  d_in[5];
  const float* nf2  = (const float*)d_in[6];
  const float* ef2  = (const float*)d_in[7];
  // d_in[8] = node_gid (implicit from contiguous layout)
  const float* x    = (const float*)d_in[9];

  FPParams P;
  P.pn_W  =(const float*)d_in[10]; P.pn_b  =(const float*)d_in[11];
  P.e1_W  =(const float*)d_in[12]; P.e1_b  =(const float*)d_in[13];
  P.e2_W  =(const float*)d_in[14]; P.e2_b  =(const float*)d_in[15];
  P.et_W  =(const float*)d_in[16]; P.et_b  =(const float*)d_in[17];
  P.g1_Wih=(const float*)d_in[18]; P.g1_bih=(const float*)d_in[19];
  P.g1_Whh=(const float*)d_in[20]; P.g1_bhh=(const float*)d_in[21];
  P.cl_W  =(const float*)d_in[22]; P.cl_b  =(const float*)d_in[23];
  P.pno_W =(const float*)d_in[24]; P.pno_b =(const float*)d_in[25];
  P.g2_Wih=(const float*)d_in[26]; P.g2_bih=(const float*)d_in[27];
  P.g2_Whh=(const float*)d_in[28]; P.g2_bhh=(const float*)d_in[29];

  HeadParams HP;
  HP.fc1_W=(const float*)d_in[30]; HP.fc1_b=(const float*)d_in[31];
  HP.fc2_W=(const float*)d_in[32]; HP.fc2_b=(const float*)d_in[33];
  HP.fc3_W=(const float*)d_in[34]; HP.fc3_b=(const float*)d_in[35];
  HP.fc4_W=(const float*)d_in[36]; HP.fc4_b=(const float*)d_in[37];
  HP.fc5_W=(const float*)d_in[38]; HP.fc5_b=(const float*)d_in[39];
  HP.fcf_W=(const float*)d_in[40]; HP.fcf_b=(const float*)d_in[41];

  float* h1 = (float*)d_ws;
  float* h2 = h1 + (size_t)NGRAPH*HF;

  afp_graph_kernel<<<NGRAPH, BLK, 0, stream>>>(src1, dst1, nf1, ef1, P, h1);
  afp_graph_kernel<<<NGRAPH, BLK, 0, stream>>>(src2, dst2, nf2, ef2, P, h2);
  head_kernel<<<(NGRAPH+255)/256, 256, 0, stream>>>(h1, h2, x, HP, (float*)d_out);
}
